// TestModule_1443109012264
// MI455X (gfx1250) — matrix operand order repaired and checked by reading, not compiled, not run on hardware
//
#include <hip/hip_runtime.h>

typedef __attribute__((ext_vector_type(16))) _Float16 v16h;
typedef __attribute__((ext_vector_type(2)))  __fp16   v2hf;  // cvt_pkrtz result type
typedef __attribute__((ext_vector_type(8)))  float    v8f;

#define H_HEADS 8
#define DHEAD   128
#define BLOCK_M 64
#define BLOCK_N 64
#define NEG_BIG (-1.0e30f)

__device__ __forceinline__ float redmax16(float x) {
  x = fmaxf(x, __shfl_xor(x, 1));
  x = fmaxf(x, __shfl_xor(x, 2));
  x = fmaxf(x, __shfl_xor(x, 4));
  x = fmaxf(x, __shfl_xor(x, 8));
  return x;
}
__device__ __forceinline__ float redsum16(float x) {
  x += __shfl_xor(x, 1);
  x += __shfl_xor(x, 2);
  x += __shfl_xor(x, 4);
  x += __shfl_xor(x, 8);
  return x;
}
__device__ __forceinline__ unsigned int h2u(v2hf h) {
  union { v2hf h; unsigned int u; } x; x.h = h; return x.u;
}

// One 64-key flash-attention step. MASKED=false is the fast interior path
// (keys guaranteed causal-valid for every row of this q-block and inside the
// segment); MASKED=true handles the diagonal band / segment tail.
template <bool MASKED>
__device__ __forceinline__ void fa_step(
    int k0,
    const float* __restrict__ kg, const float* __restrict__ vg,
    int head, int kbeg, int kend, int kmax, int qbeg, int qrowM,
    int kpr, int dcol, int hl, int nn, int wv,
    _Float16* sK, _Float16* sVT, _Float16* sPh,
    const v16h (&qa)[4], v8f (&oacc)[8], float (&mrun)[8], float (&lrun)[8])
{
  __syncthreads();  // previous step's LDS reads complete

  // ---- stage K (row-major f16) and V (transposed f16) into LDS ----
  // Each thread: key rows {2*kpr, 2*kpr+1}, d-chunk [dcol, dcol+32).
  const int r0 = 2 * kpr, r1 = r0 + 1;
  const int g0 = k0 + r0, g1 = k0 + r1;
  bool ok0 = true, ok1 = true;
  if (MASKED) { ok0 = g0 < kend; ok1 = g1 < kend; }
  const int gc0 = ok0 ? g0 : kbeg;
  const int gc1 = ok1 ? g1 : kbeg;
  const float* kp0 = kg + ((size_t)gc0 * H_HEADS + head) * DHEAD + dcol;
  const float* kp1 = kg + ((size_t)gc1 * H_HEADS + head) * DHEAD + dcol;
  const float* vp0 = vg + ((size_t)gc0 * H_HEADS + head) * DHEAD + dcol;
  const float* vp1 = vg + ((size_t)gc1 * H_HEADS + head) * DHEAD + dcol;
  #pragma unroll
  for (int j = 0; j < 8; ++j) {
    const int d = dcol + j * 4;
    float4 fk0 = *(const float4*)(kp0 + j * 4);
    float4 fk1 = *(const float4*)(kp1 + j * 4);
    float4 fv0 = *(const float4*)(vp0 + j * 4);
    float4 fv1 = *(const float4*)(vp1 + j * 4);
    if (MASKED && !ok0) { fk0.x=fk0.y=fk0.z=fk0.w=0.f; fv0.x=fv0.y=fv0.z=fv0.w=0.f; }
    if (MASKED && !ok1) { fk1.x=fk1.y=fk1.z=fk1.w=0.f; fv1.x=fv1.y=fv1.z=fv1.w=0.f; }
    // K rows: pack along d -> one b64 store per row per j
    uint2 w0; w0.x = h2u(__builtin_amdgcn_cvt_pkrtz(fk0.x, fk0.y));
              w0.y = h2u(__builtin_amdgcn_cvt_pkrtz(fk0.z, fk0.w));
    uint2 w1; w1.x = h2u(__builtin_amdgcn_cvt_pkrtz(fk1.x, fk1.y));
              w1.y = h2u(__builtin_amdgcn_cvt_pkrtz(fk1.z, fk1.w));
    *(uint2*)&sK[r0 * DHEAD + d] = w0;
    *(uint2*)&sK[r1 * DHEAD + d] = w1;
    // V transposed: adjacent keys are adjacent in sVT -> packed b32 stores
    *(unsigned int*)&sVT[(d + 0) * BLOCK_N + r0] = h2u(__builtin_amdgcn_cvt_pkrtz(fv0.x, fv1.x));
    *(unsigned int*)&sVT[(d + 1) * BLOCK_N + r0] = h2u(__builtin_amdgcn_cvt_pkrtz(fv0.y, fv1.y));
    *(unsigned int*)&sVT[(d + 2) * BLOCK_N + r0] = h2u(__builtin_amdgcn_cvt_pkrtz(fv0.z, fv1.z));
    *(unsigned int*)&sVT[(d + 3) * BLOCK_N + r0] = h2u(__builtin_amdgcn_cvt_pkrtz(fv0.w, fv1.w));
  }
  if (k0 + BLOCK_N + r0 < kmax) {  // next tile -> global_prefetch_b8
    const size_t nb = ((size_t)(k0 + BLOCK_N + r0) * H_HEADS + head) * DHEAD + dcol;
    __builtin_prefetch(kg + nb, 0, 0);
    __builtin_prefetch(vg + nb, 0, 0);
  }
  __syncthreads();

  // ---- S = Q @ K^T : four 16x16 N-tiles, 4 K-chunks each (16 WMMA) ----
  v8f acc[4];
  #pragma unroll
  for (int nt = 0; nt < 4; ++nt) acc[nt] = v8f{0.f,0.f,0.f,0.f,0.f,0.f,0.f,0.f};
  #pragma unroll
  for (int c = 0; c < 4; ++c) {
    #pragma unroll
    for (int nt = 0; nt < 4; ++nt) {
      v16h b = *(const v16h*)&sK[(nt * 16 + nn) * DHEAD + c * 32 + hl * 16];
      acc[nt] = __builtin_amdgcn_wmma_f32_16x16x32_f16(false, qa[c], false, b,
                                                       (short)0, acc[nt], false, false);
    }
  }

  // ---- online softmax in exp2 domain; P written to LDS in A-layout order ----
  int ppos[4];
  #pragma unroll
  for (int nt = 0; nt < 4; ++nt) {
    const int kk = nt * 16 + nn;
    const int kp32 = kk & 31;
    ppos[nt] = (kk >> 5) * 32 + kp32;
  }
  v8f av;
  #pragma unroll
  for (int r = 0; r < 8; ++r) {
    float sv[4];
    #pragma unroll
    for (int nt = 0; nt < 4; ++nt) sv[nt] = acc[nt][r];
    if (MASKED) {
      const int qi = qrowM + hl * 8 + r;
      const int pq = qi - qbeg;
      #pragma unroll
      for (int nt = 0; nt < 4; ++nt) {
        const int ki = k0 + nt * 16 + nn;
        sv[nt] = ((ki < kend) && ((ki - kbeg) <= pq)) ? sv[nt] : NEG_BIG;
      }
    }
    float mx = redmax16(fmaxf(fmaxf(sv[0], sv[1]), fmaxf(sv[2], sv[3])));
    mx = fmaxf(mx, mrun[r]);
    const float al = exp2f(mrun[r] - mx);
    mrun[r] = mx;
    float p[4];
    #pragma unroll
    for (int nt = 0; nt < 4; ++nt) p[nt] = exp2f(sv[nt] - mx);
    lrun[r] = lrun[r] * al + redsum16((p[0] + p[1]) + (p[2] + p[3]));
    av[r] = al;
    const int rowb = wv * 1024 + (hl * 8 + r) * BLOCK_N;
    #pragma unroll
    for (int nt = 0; nt < 4; ++nt)
      sPh[rowb + ppos[nt]] = (_Float16)p[nt];
  }
  #pragma unroll
  for (int dt = 0; dt < 8; ++dt) oacc[dt] *= av;

  __syncthreads();  // sPh visible across lanes; orders sVT reads vs next stage

  // ---- P A-fragments: one contiguous v16h per 32-key chunk ----
  v16h pa[2];
  #pragma unroll
  for (int part = 0; part < 2; ++part)
    pa[part] = *(const v16h*)&sPh[wv * 1024 + nn * BLOCK_N + part * 32 + hl * 16];

  // ---- O += P @ V : 8 d-tiles x 2 key-chunks (16 WMMA) ----
  #pragma unroll
  for (int dt = 0; dt < 8; ++dt) {
    #pragma unroll
    for (int part = 0; part < 2; ++part) {
      v16h vb = *(const v16h*)&sVT[(dt * 16 + nn) * BLOCK_N + part * 32 + hl * 16];
      oacc[dt] = __builtin_amdgcn_wmma_f32_16x16x32_f16(false, pa[part], false, vb,
                                                        (short)0, oacc[dt], false, false);
    }
  }
}

__global__ __launch_bounds__(128)
void fa_varlen_fwd(const float* __restrict__ q,
                   const float* __restrict__ k,
                   const float* __restrict__ v,
                   const int* __restrict__ cu_q,
                   const int* __restrict__ cu_k,
                   float* __restrict__ out,
                   int total_q, int total_k, int nseq)
{
  __shared__ alignas(32) _Float16 sK [BLOCK_N * DHEAD];   // 16 KB K tile
  __shared__ alignas(32) _Float16 sVT[DHEAD * BLOCK_N];   // 16 KB V^T tile
  __shared__ alignas(32) _Float16 sPh[4 * 16 * BLOCK_N];  //  8 KB per-wave P

  const int tid  = threadIdx.x;
  const int lane = tid & 31;
  const int wv   = tid >> 5;
  const int hl   = lane >> 4;
  const int nn   = lane & 15;
  const int head = blockIdx.y;
  const int m0   = blockIdx.x * BLOCK_M;
  if (m0 >= total_q) return;

  // ---- segment lookup (uniform per workgroup) ----
  int seg = 0;
  for (int j = 1; j < nseq; ++j) if (cu_q[j] <= m0) seg = j;
  const int qbeg = cu_q[seg], qend = cu_q[seg + 1];
  const int kbeg = cu_k[seg], kend = cu_k[seg + 1];
  int maxq = m0 + BLOCK_M - 1;
  if (maxq > qend - 1)    maxq = qend - 1;
  if (maxq > total_q - 1) maxq = total_q - 1;
  int kmax = kbeg + (maxq - qbeg) + 1;
  if (kmax > kend)    kmax = kend;
  if (kmax > total_k) kmax = total_k;
  // keys valid for EVERY row of this q-block (full 64-key blocks only)
  int kfull = kbeg + (((m0 - qbeg) + 1) & ~(BLOCK_N - 1));
  if (kfull > kmax) kfull = kmax;

  // 1/sqrt(128) * log2(e): softmax runs in exp2 domain
  const float qscale = 0.08838834764831845f * 1.4426950408889634f;

  // ---- Q A-fragments ----
  const int qrowM = m0 + wv * 16;
  int qiA = qrowM + nn; if (qiA > total_q - 1) qiA = total_q - 1;
  const float* qp = q + ((size_t)qiA * H_HEADS + head) * DHEAD;
  v16h qa[4];
  #pragma unroll
  for (int c = 0; c < 4; ++c) {
    v16h a;
    #pragma unroll
    for (int rs = 0; rs < 2; ++rs) {
      const int db = c * 32 + hl * 16 + rs * 8;
      float4 f0 = *(const float4*)(qp + db);
      float4 f1 = *(const float4*)(qp + db + 4);
      const int e = rs * 8;
      a[e + 0] = (_Float16)(f0.x * qscale); a[e + 1] = (_Float16)(f0.y * qscale);
      a[e + 2] = (_Float16)(f0.z * qscale); a[e + 3] = (_Float16)(f0.w * qscale);
      a[e + 4] = (_Float16)(f1.x * qscale); a[e + 5] = (_Float16)(f1.y * qscale);
      a[e + 6] = (_Float16)(f1.z * qscale); a[e + 7] = (_Float16)(f1.w * qscale);
    }
    qa[c] = a;
  }

  v8f oacc[8];
  #pragma unroll
  for (int dt = 0; dt < 8; ++dt) oacc[dt] = v8f{0.f,0.f,0.f,0.f,0.f,0.f,0.f,0.f};
  float mrun[8], lrun[8];
  #pragma unroll
  for (int r = 0; r < 8; ++r) { mrun[r] = NEG_BIG; lrun[r] = 0.f; }

  const int kpr  = tid >> 2;        // key-pair index 0..31
  const int dcol = (tid & 3) * 32;  // d-chunk

  int k0 = kbeg;
  for (; k0 < kfull; k0 += BLOCK_N)   // interior: no masking
    fa_step<false>(k0, k, v, head, kbeg, kend, kmax, qbeg, qrowM,
                   kpr, dcol, hl, nn, wv, sK, sVT, sPh, qa, oacc, mrun, lrun);
  for (; k0 < kmax; k0 += BLOCK_N)    // diagonal band / tail: masked
    fa_step<true>(k0, k, v, head, kbeg, kend, kmax, qbeg, qrowM,
                  kpr, dcol, hl, nn, wv, sK, sVT, sPh, qa, oacc, mrun, lrun);

  // ---- epilogue: normalize and store ----
  #pragma unroll
  for (int r = 0; r < 8; ++r) {
    const int qi = qrowM + hl * 8 + r;
    if (qi < total_q) {
      const float li = 1.0f / lrun[r];
      float* op = out + ((size_t)qi * H_HEADS + head) * DHEAD;
      #pragma unroll
      for (int dt = 0; dt < 8; ++dt)
        op[dt * 16 + nn] = oacc[dt][r] * li;
    }
  }
}

extern "C" void kernel_launch(void* const* d_in, const int* in_sizes, int n_in,
                              void* d_out, int out_size, void* d_ws, size_t ws_size,
                              hipStream_t stream) {
  (void)n_in; (void)out_size; (void)d_ws; (void)ws_size;
  const float* q    = (const float*)d_in[0];
  const float* k    = (const float*)d_in[1];
  const float* v    = (const float*)d_in[2];
  const int*   cu_q = (const int*)d_in[3];
  const int*   cu_k = (const int*)d_in[4];
  float*       out  = (float*)d_out;

  const int total_q = in_sizes[0] / (H_HEADS * DHEAD);
  const int total_k = in_sizes[1] / (H_HEADS * DHEAD);
  const int nseq    = in_sizes[3] - 1;

  dim3 grid((total_q + BLOCK_M - 1) / BLOCK_M, H_HEADS);
  fa_varlen_fwd<<<grid, 128, 0, stream>>>(q, k, v, cu_q, cu_k, out,
                                          total_q, total_k, nseq);
}
